// GCN_46703474376725
// MI455X (gfx1250) — matrix operand order repaired and checked by reading, not compiled, not run on hardware
//
#include <hip/hip_runtime.h>

// ---------------------------------------------------------------------------
// 2-layer GCN for MI455X (gfx1250), wave32 + WMMA.
//   dis = rsqrt(deg)          (deg = 1 + indegree, self loops folded in)
//   h1  = X @ W1              (v_wmma_f32_16x16x32_f16, f32 accumulate)
//   a1  = relu(scatter + h1*dis^2 + b1)      -> stored f16 for next GEMM
//   h2  = a1 @ W2             (WMMA)
//   out = scatter + h2*dis^2 + b2
// ---------------------------------------------------------------------------

typedef __attribute__((ext_vector_type(16))) _Float16 v16h;
typedef __attribute__((ext_vector_type(8)))  _Float16 v8h;
typedef __attribute__((ext_vector_type(8)))  float    v8f;

#define HID 128

// ---------------- small utility kernels ----------------

__global__ void fill_f32(float* p, float v, long long n) {
    long long i = blockIdx.x * (long long)blockDim.x + threadIdx.x;
    if (i < n) p[i] = v;
}

__global__ void deg_accum(float* deg, const long long* __restrict__ dst, long long E) {
    long long e = blockIdx.x * (long long)blockDim.x + threadIdx.x;
    if (e < E) atomicAdd(&deg[dst[e]], 1.0f);
}

__global__ void deg_rsqrt(float* deg, int n) {
    int i = blockIdx.x * blockDim.x + threadIdx.x;
    if (i < n) deg[i] = rsqrtf(deg[i]);   // deg >= 1 always (self loop)
}

// f32 -> f16 elementwise, 8 elements per thread (vectorized b128 in / b128 out)
__global__ void cvt_f32_to_f16(const float* __restrict__ in, _Float16* __restrict__ out,
                               long long n8) {
    long long i = blockIdx.x * (long long)blockDim.x + threadIdx.x;
    if (i >= n8) return;
    const float4 f0 = *(const float4*)(in + i * 8);
    const float4 f1 = *(const float4*)(in + i * 8 + 4);
    v8h o;
    o[0] = (_Float16)f0.x; o[1] = (_Float16)f0.y; o[2] = (_Float16)f0.z; o[3] = (_Float16)f0.w;
    o[4] = (_Float16)f1.x; o[5] = (_Float16)f1.y; o[6] = (_Float16)f1.z; o[7] = (_Float16)f1.w;
    *(v8h*)(out + i * 8) = o;
}

// Convert W[K,N] (f32 row-major) into per-wave WMMA B fragments, kt-major:
// frag[((kt*8 + nt)*32 + lane)*16 + j] = W[kt*32 + (lane>>4)*16 + j][nt*16 + (lane&15)]
// => inside the GEMM loop the 8 per-kt fragments sit at consecutive 1KB strides.
__global__ void convert_wfrag(const float* __restrict__ W, _Float16* __restrict__ frag,
                              int K, int N) {
    int ktiles = K >> 5;
    int total  = ktiles * (N >> 4) * 32;
    int tid = blockIdx.x * blockDim.x + threadIdx.x;
    if (tid >= total) return;
    int lane  = tid & 31;
    int group = tid >> 5;
    int nt    = group & 7;        // N/16 == 8
    int kt    = group >> 3;
    int n     = nt * 16 + (lane & 15);
    int kb    = kt * 32 + ((lane >> 4) << 4);
    _Float16* dp = frag + (size_t)tid * 16;
#pragma unroll
    for (int j = 0; j < 16; ++j) dp[j] = (_Float16)W[(size_t)(kb + j) * N + n];
}

// ---------------- WMMA GEMM: C[M,HID] = A16[M,K] @ B[K,HID] ----------------
// 256 threads = 8 waves; each wave owns one 16-row tile across all 128 columns
// (8 accumulator tiles). Per K-step: 2 A loads + 8 B loads (immediate offsets)
// + 8 WMMAs, with B preloaded into registers for load/compute overlap.
__global__ __launch_bounds__(256) void gemm_wmma_f16(
    const _Float16* __restrict__ A16, const _Float16* __restrict__ Bfrag,
    float* __restrict__ C, int M, int K) {
    const int lane = threadIdx.x & 31;
    const int wave = threadIdx.x >> 5;
    const int tile = blockIdx.x * 8 + wave;     // 16-row tile index
    const int m0   = tile << 4;
    if (m0 >= M) return;                        // M is a multiple of 16 (50000/16 = 3125)
    const int ktiles = K >> 5;
    const int mrow   = m0 + (lane & 15);
    const int kb8    = (lane >> 4) << 4;

    v8f acc[8];
#pragma unroll
    for (int n = 0; n < 8; ++n) acc[n] = (v8f){0.f,0.f,0.f,0.f,0.f,0.f,0.f,0.f};

    const _Float16* aptr = A16 + (size_t)mrow * K + kb8;
    const _Float16* bptr = Bfrag + (size_t)lane * 16;   // + kt*4096 + n*512 (halves)

    for (int kt = 0; kt < ktiles; ++kt) {
        // A fragment: halves 0..15 = K[kb8..kb8+15]
        v8h alo = *(const v8h*)(aptr);
        v8h ahi = *(const v8h*)(aptr + 8);
        v16h a = __builtin_shufflevector(alo, ahi,
                                         0, 1, 2, 3, 4, 5, 6, 7,
                                         8, 9, 10, 11, 12, 13, 14, 15);
        v16h bf[8];
#pragma unroll
        for (int n = 0; n < 8; ++n)
            bf[n] = *(const v16h*)(bptr + n * 512);
#pragma unroll
        for (int n = 0; n < 8; ++n)
            acc[n] = __builtin_amdgcn_wmma_f32_16x16x32_f16(
                false, a, false, bf[n], (short)0, acc[n], false, false);
        aptr += 32;
        bptr += 4096;
    }

    const int ncol  = lane & 15;
    const int mbase = m0 + ((lane >> 4) << 3);
#pragma unroll
    for (int n = 0; n < 8; ++n) {
        float* cp = C + (size_t)mbase * HID + (n << 4) + ncol;
#pragma unroll
        for (int r = 0; r < 8; ++r) cp[(size_t)r * HID] = acc[n][r];
    }
}

// ---------------- edge aggregation (scatter-add) ----------------
// One thread per (edge, 4-column group): 32 threads cover 128 features.
__global__ void agg_edges(const float* __restrict__ h,
                          const long long* __restrict__ src,
                          const long long* __restrict__ dst,
                          const float* __restrict__ dis,
                          float* out, long long E) {
    long long idx = blockIdx.x * (long long)blockDim.x + threadIdx.x;
    if (idx >= E * 32) return;
    long long e = idx >> 5;
    int c4 = (int)(idx & 31) << 2;
    long long s = src[e], d = dst[e];
    float w = dis[s] * dis[d];
    float4 hv = *(const float4*)(h + s * HID + c4);
    float* op = out + d * HID + c4;
    atomicAdd(op + 0, hv.x * w);
    atomicAdd(op + 1, hv.y * w);
    atomicAdd(op + 2, hv.z * w);
    atomicAdd(op + 3, hv.w * w);
}

// out16[i][c] = (f16) relu(agg[i][c] + h[i][c]*dis[i]^2 + b[c])  -- feeds GEMM2
__global__ void finalize_relu_f16(const float* __restrict__ agg,
                                  const float* __restrict__ h,
                                  const float* __restrict__ dis,
                                  const float* __restrict__ b,
                                  _Float16* __restrict__ out16, long long n) {
    long long idx = blockIdx.x * (long long)blockDim.x + threadIdx.x;
    if (idx >= n) return;
    int i = (int)(idx >> 7), c = (int)(idx & (HID - 1));
    float ds = dis[i];
    float v = agg[idx] + h[idx] * ds * ds + b[c];
    out16[idx] = (_Float16)(v > 0.f ? v : 0.f);
}

// out[i][c] += h[i][c]*dis[i]^2 + b[c]   (no relu on layer 2)
__global__ void finalize_out(float* out, const float* __restrict__ h,
                             const float* __restrict__ dis,
                             const float* __restrict__ b, long long n) {
    long long idx = blockIdx.x * (long long)blockDim.x + threadIdx.x;
    if (idx >= n) return;
    int i = (int)(idx >> 7), c = (int)(idx & (HID - 1));
    float ds = dis[i];
    out[idx] = out[idx] + h[idx] * ds * ds + b[c];
}

// ---------------- host orchestration ----------------

static inline size_t align256(size_t x) { return (x + 255) & ~(size_t)255; }

extern "C" void kernel_launch(void* const* d_in, const int* in_sizes, int n_in,
                              void* d_out, int out_size, void* d_ws, size_t ws_size,
                              hipStream_t stream) {
    const float*     x   = (const float*)d_in[0];
    const long long* ei  = (const long long*)d_in[1];
    const float*     W1  = (const float*)d_in[2];
    const float*     b1  = (const float*)d_in[3];
    const float*     W2  = (const float*)d_in[4];
    const float*     b2  = (const float*)d_in[5];
    float*           out = (float*)d_out;

    const int       K1  = 256, K2 = HID;
    const int       Nn  = in_sizes[0] / K1;       // 50000 nodes
    const long long E   = in_sizes[1] / 2;        // 1.6M edges
    const long long NH  = (long long)Nn * HID;

    const long long* src = ei;
    const long long* dst = ei + E;

    // workspace carve-up (~90 MB)
    char* w = (char*)d_ws;
    float*    dis    = (float*)w;            w += align256((size_t)Nn * 4);
    _Float16* W1h    = (_Float16*)w;         w += align256((size_t)K1 * HID * 2);
    _Float16* W2h    = (_Float16*)w;         w += align256((size_t)K2 * HID * 2);
    _Float16* x16    = (_Float16*)w;         w += align256((size_t)Nn * K1 * 2);
    _Float16* a16    = (_Float16*)w;         w += align256((size_t)NH * 2);
    float*    hbuf   = (float*)w;            w += align256((size_t)NH * 4);
    float*    aggbuf = (float*)w;            w += align256((size_t)NH * 4);

    const int T = 256;
    dim3 blk(T);

    // 1) degrees (self loop => init 1.0), then dis = rsqrt(deg)
    fill_f32<<<dim3((Nn + T - 1) / T), blk, 0, stream>>>(dis, 1.0f, Nn);
    deg_accum<<<dim3((unsigned)((E + T - 1) / T)), blk, 0, stream>>>(dis, dst, E);
    deg_rsqrt<<<dim3((Nn + T - 1) / T), blk, 0, stream>>>(dis, Nn);

    // 2) f16 operands: weight fragments + converted X
    {
        int tot1 = (K1 / 32) * (HID / 16) * 32;
        int tot2 = (K2 / 32) * (HID / 16) * 32;
        convert_wfrag<<<dim3((tot1 + T - 1) / T), blk, 0, stream>>>(W1, W1h, K1, HID);
        convert_wfrag<<<dim3((tot2 + T - 1) / T), blk, 0, stream>>>(W2, W2h, K2, HID);
        long long n8 = (long long)Nn * K1 / 8;
        cvt_f32_to_f16<<<dim3((unsigned)((n8 + T - 1) / T)), blk, 0, stream>>>(x, x16, n8);
    }

    const unsigned gemmBlocks = (unsigned)((Nn / 16 + 7) / 8);    // 391
    const unsigned aggBlocks  = (unsigned)((E * 32 + T - 1) / T); // 200000
    const unsigned nhBlocks   = (unsigned)((NH + T - 1) / T);

    // 3) layer 1: h1 = X @ W1
    gemm_wmma_f16<<<dim3(gemmBlocks), blk, 0, stream>>>(x16, W1h, hbuf, Nn, K1);

    // 4) aggregate into aggbuf, fold in self loop + bias + relu, emit f16
    fill_f32<<<dim3(nhBlocks), blk, 0, stream>>>(aggbuf, 0.0f, NH);
    agg_edges<<<dim3(aggBlocks), blk, 0, stream>>>(hbuf, src, dst, dis, aggbuf, E);
    finalize_relu_f16<<<dim3(nhBlocks), blk, 0, stream>>>(aggbuf, hbuf, dis, b1, a16, NH);

    // 5) layer 2: h2 = a1 @ W2
    gemm_wmma_f16<<<dim3(gemmBlocks), blk, 0, stream>>>(a16, W2h, hbuf, Nn, K2);

    // 6) aggregate into d_out, fold in self loop + bias
    fill_f32<<<dim3(nhBlocks), blk, 0, stream>>>(out, 0.0f, NH);
    agg_edges<<<dim3(aggBlocks), blk, 0, stream>>>(hbuf, src, dst, dis, out, E);
    finalize_out<<<dim3(nhBlocks), blk, 0, stream>>>(out, hbuf, dis, b2, NH);
}
